// istsplm_forecast_33861522162499
// MI455X (gfx1250) — matrix operand order repaired and checked by reading, not compiled, not run on hardware
//
#include <hip/hip_runtime.h>
#include <cstddef>
#include <cstdint>

// ---------------------------------------------------------------------------
// ISTS-PLM forecast forward for MI455X (gfx1250, wave32, WMMA + TDM).
// All dense GEMMs go through v_wmma_f32_16x16x32_bf16 (bf16 in, f32 acc).
// A (activation) tiles are DMA'd into LDS by the Tensor Data Mover
// (tensor_load_to_lds, gated by s_wait_tensorcnt); weight tiles are staged
// with vectorized b128 loads + transpose scatter into LDS.
// ---------------------------------------------------------------------------

using bf16 = __bf16;
typedef __attribute__((ext_vector_type(16))) __bf16 bf16x16;
typedef __attribute__((ext_vector_type(8)))  float  floatx8;
typedef __attribute__((ext_vector_type(4)))  unsigned int u32x4;
typedef __attribute__((ext_vector_type(8)))  int          i32x8;
typedef __attribute__((ext_vector_type(4)))  int          i32x4;

#define NEG_INF (-3.4028234663852886e38f)

// Model constants
#define D_VARS   12
#define BATCH    2
#define L_OBS    512
#define L_PRED   96
#define D_MODEL  1024
#define N_HEADS  16
#define N_KV     8
#define HD       64
#define FF       4096
#define PROMPT   8
#define S_A      (PROMPT + L_OBS)          // 520
#define M_A      (BATCH * D_VARS * S_A)    // 12480
#define M_B      (BATCH * D_VARS)          // 24
#define S_B      D_VARS                    // 12
#define MLP_CHUNK 1560                     // 8 chunks of stage-A rows

// ---------------------------------------------------------------------------
// f32 -> bf16 convert (weights, activations)
// ---------------------------------------------------------------------------
__global__ void k_f2b(const float* __restrict__ in, bf16* __restrict__ out, size_t n) {
    size_t i = (size_t)blockIdx.x * blockDim.x + threadIdx.x;
    if (i < n) out[i] = (bf16)in[i];
}

// ---------------------------------------------------------------------------
// Stage-A input build: prompt/value embedding, time vector, additive mask
// ---------------------------------------------------------------------------
__global__ __launch_bounds__(256)
void k_build_inputs(const float* __restrict__ od, const float* __restrict__ om,
                    const float* __restrict__ otp, const float* __restrict__ val_w,
                    const float* __restrict__ val_b, const float* __restrict__ prompt,
                    float* __restrict__ x, float* __restrict__ tt, float* __restrict__ madd) {
    int row = blockIdx.x;                // bd*S_A + s
    int bd = row / S_A, s = row % S_A;
    int b = bd / D_VARS, dv = bd % D_VARS;
    if (threadIdx.x == 0) {
        float t = (s < PROMPT) ? 0.f : otp[b * L_OBS + (s - PROMPT)];
        tt[row] = t;
        float m = (s < PROMPT) ? 1.f : om[((size_t)b * L_OBS + (s - PROMPT)) * D_VARS + dv];
        madd[row] = (1.f - m) * NEG_INF;
    }
    if (s < PROMPT) {
        for (int c = threadIdx.x; c < D_MODEL; c += 256)
            x[(size_t)row * D_MODEL + c] = prompt[s * D_MODEL + c];
    } else {
        float xv = od[((size_t)b * L_OBS + (s - PROMPT)) * D_VARS + dv];
        for (int c = threadIdx.x; c < D_MODEL; c += 256)
            x[(size_t)row * D_MODEL + c] = xv * val_w[c] + val_b[c];
    }
}

// ---------------------------------------------------------------------------
// RMSNorm: one block per row; writes bf16 (for GEMM A) and/or f32
// ---------------------------------------------------------------------------
__global__ __launch_bounds__(256)
void k_rmsnorm(const float* __restrict__ in, const float* __restrict__ w,
               bf16* __restrict__ outb, float* __restrict__ outf, int D) {
    int row = blockIdx.x;
    const float* xr = in + (size_t)row * D;
    __shared__ float red[256];
    float s = 0.f;
    for (int c = threadIdx.x; c < D; c += 256) { float v = xr[c]; s += v * v; }
    red[threadIdx.x] = s; __syncthreads();
    for (int st = 128; st > 0; st >>= 1) {
        if (threadIdx.x < st) red[threadIdx.x] += red[threadIdx.x + st];
        __syncthreads();
    }
    float inv = rsqrtf(red[0] / (float)D + 1e-6f);
    for (int c = threadIdx.x; c < D; c += 256) {
        float v = xr[c] * inv * w[c];
        if (outb) outb[(size_t)row * D + c] = (bf16)v;
        if (outf) outf[(size_t)row * D + c] = v;
    }
}

// ---------------------------------------------------------------------------
// Tiled WMMA GEMM: C[M,N] = A[M,K](bf16) @ W[K,N](bf16)  (+bias)(+act)(+res)
// 256 threads = 8 wave32 waves; block tile 128x64; K stepped by 32.
// A tile is loaded by the Tensor Data Mover (wave 0 issues the descriptor,
// completion gated by s_wait_tensorcnt + workgroup barrier); rows past M
// read as zero via D# OOB semantics. Fragment layouts per ISA 7.12.2.
// mode: 0 plain, 1 silu(aux)*acc, 2 leaky_relu(acc+bias)
// ---------------------------------------------------------------------------
#define GT_M 128
#define GT_N 64
#define GT_K 32

// Issue a 2D TDM load: rows x 32 bf16 elements, row stride = K elements,
// from global 'gptr' into LDS offset 'lds_addr' (packed [rows][32]).
__device__ __forceinline__ void tdm_load_a_tile(const bf16* gptr, unsigned lds_addr,
                                                int rows_in_bounds, int K) {
    unsigned long long ga = (unsigned long long)(uintptr_t)gptr;
    // ---- D# group 0 (ISA 8.3): count=1, lds_addr, global_addr, type=2 ----
    u32x4 g0;
    g0[0] = 1u;                                   // count=1 (valid), user mode
    g0[1] = lds_addr;                             // LDS byte address
    g0[2] = (unsigned)(ga & 0xFFFFFFFFu);         // global_addr[31:0]
    g0[3] = (unsigned)((ga >> 32) & 0x01FFFFFFu)  // global_addr[56:32]
          | (2u << 30);                           // type=2 ("image")
    // ---- D# group 1 (ISA 8.4) ----
    unsigned td0 = (unsigned)GT_K;                // tensor_dim0 (X extent for OOB)
    unsigned td1 = (unsigned)(rows_in_bounds > 0 ? rows_in_bounds : 0); // rows
    unsigned long long s0 = (unsigned long long)K; // dim0 stride (elements)
    i32x8 g1;
    g1[0] = (int)(1u << 16);                      // data_size=1 (2 bytes/elem)
    g1[1] = (int)((td0 & 0xFFFFu) << 16);         // tensor_dim0[15:0] @ bits 63:48
    g1[2] = (int)(((td0 >> 16) & 0xFFFFu) | ((td1 & 0xFFFFu) << 16));
    g1[3] = (int)(((td1 >> 16) & 0xFFFFu) | ((unsigned)GT_K << 16)); // tile_dim0=32
    g1[4] = (int)(unsigned)GT_M;                  // tile_dim1=128, tile_dim2=0
    g1[5] = (int)(unsigned)(s0 & 0xFFFFFFFFu);    // tensor_dim0_stride[31:0]
    g1[6] = (int)(unsigned)((s0 >> 32) & 0xFFFFu);
    g1[7] = 0;
    i32x4 gz; gz[0] = 0; gz[1] = 0; gz[2] = 0; gz[3] = 0;   // groups 2/3: 2D tensor
#if __has_include(<hip/amd_detail/amd_gfx1250_TDM.h>)
    i32x8 gz8;
    for (int i = 0; i < 8; ++i) gz8[i] = 0;
    __builtin_amdgcn_tensor_load_to_lds(g0, g1, gz, gz, gz8, 0);   // clang-23 form
#else
    __builtin_amdgcn_tensor_load_to_lds(g0, g1, gz, gz, 0);        // ROCm 7.2 form
#endif
}

__global__ __launch_bounds__(256)
void k_gemm(const bf16* __restrict__ A, const bf16* __restrict__ Wm,
            const float* __restrict__ bias, const float* __restrict__ res,
            const float* __restrict__ aux,
            float* __restrict__ Cf, bf16* __restrict__ Cb,
            int M, int N, int K, int mode) {
    __shared__ bf16 As[GT_M][GT_K];   // 8 KB, filled by TDM
    __shared__ bf16 Bs[GT_N][GT_K];   // 4 KB, W tile stored transposed [n][k]
    const int tid  = threadIdx.x;
    const int wave = tid >> 5;
    const int lane = tid & 31;
    const int m0 = blockIdx.y * GT_M;
    const int n0 = blockIdx.x * GT_N;

    floatx8 acc[4];
    for (int c = 0; c < 4; ++c)
        for (int i = 0; i < 8; ++i) acc[c][i] = 0.f;

    const int b_k  = tid & 31;              // k within tile
    const int b_ng = tid >> 5;              // 8-column group
    const bool n_full = (n0 + GT_N <= N);   // loop-invariant tail predicate
    // Low 32 bits of a flat shared pointer are the LDS byte address (ISA 10.2).
    const unsigned as_lds = (unsigned)(uintptr_t)(&As[0][0]);

    for (int kk = 0; kk < K; kk += GT_K) {
        // --- A tile via Tensor Data Mover (wave 0 only issues the DMA) ---
        if (wave == 0) {
            tdm_load_a_tile(A + (size_t)m0 * K + kk, as_lds, M - m0, K);
        }
        // --- B tile: W rows kk..kk+31, cols n0..n0+63, transposed into LDS ---
        {
            const int gk = kk + b_k;
            const bf16* wrow = Wm + (size_t)gk * N + n0 + b_ng * 8;
            if (n_full) {
                // single global_load_b128 (16B aligned: N multiple of 8 here)
                union { uint4 q; bf16 e[8]; } u;
                u.q = *(const uint4*)wrow;
                for (int j = 0; j < 8; ++j)
                    Bs[b_ng * 8 + j][b_k] = u.e[j];
            } else {
                // tail (head GEMM, N=12): clamped address + select, no branches
                for (int j = 0; j < 8; ++j) {
                    int gn = n0 + b_ng * 8 + j;
                    int gc = gn < N ? gn : N - 1;
                    bf16 v = Wm[(size_t)gk * N + gc];
                    Bs[b_ng * 8 + j][b_k] = (gn < N) ? v : (bf16)0.f;
                }
            }
            if (kk + GT_K < K)   // speculative prefetch of next W slab
                __builtin_prefetch(Wm + (size_t)(kk + GT_K + b_k) * N + n0, 0, 1);
        }
        if (wave == 0) __builtin_amdgcn_s_wait_tensorcnt(0);
        __syncthreads();

        // --- build fragments & issue 4 WMMAs (16x16x32 bf16 -> f32) ---
        union Frag { bf16x16 v; unsigned u[8]; };
        const int fm = lane & 15;
        const int kh = lane >> 4;   // K-half select
        Frag af;
        {
            const unsigned* arow = (const unsigned*)&As[wave * 16 + fm][0];
            for (int r = 0; r < 4; ++r) {
                af.u[r]     = arow[4 * kh + r];       // K = 8*kh + 2r (+1)
                af.u[4 + r] = arow[8 + 4 * kh + r];   // K = 16 + 8*kh + 2r (+1)
            }
        }
        for (int c = 0; c < 4; ++c) {
            Frag bfr;
            const unsigned* brow = (const unsigned*)&Bs[c * 16 + fm][0];
            for (int j = 0; j < 8; ++j) bfr.u[j] = brow[(j < 4 ? 4 * kh : 4 + 4 * kh) + j];  // K = 8*kh + 2j (+1), j<4; 16 + 8*kh + 2(j-4) (+1), j>=4
            acc[c] = __builtin_amdgcn_wmma_f32_16x16x32_bf16(
                false, af.v, false, bfr.v, (short)0, acc[c], false, false);
        }
        __syncthreads();
    }

    // --- epilogue: D element (m,n) -> lane = n + 16*(m/8), vgpr = m%8 ---
    const int n_l = lane & 15;
    const int kh  = lane >> 4;
    for (int c = 0; c < 4; ++c) {
        int gn = n0 + c * 16 + n_l;
        if (gn >= N) continue;
        for (int r = 0; r < 8; ++r) {
            int gm = m0 + wave * 16 + kh * 8 + r;
            if (gm >= M) continue;
            float val = acc[c][r];
            if (bias) val += bias[gn];
            size_t off = (size_t)gm * N + gn;
            if (mode == 1) {                       // silu(aux) * acc
                float g = aux[off];
                val = (g / (1.f + expf(-g))) * val;
            } else if (mode == 2) {                // leaky relu
                val = val > 0.f ? val : 0.01f * val;
            }
            if (res) val += res[off];
            if (Cf) Cf[off] = val;
            if (Cb) Cb[off] = (bf16)val;
        }
    }
}

// ---------------------------------------------------------------------------
// RoPE in place on q[M,16,64] / k[M,8,64], t = tt[row]
// ---------------------------------------------------------------------------
__global__ void k_rope(float* __restrict__ q, float* __restrict__ k,
                       const float* __restrict__ tt, int M) {
    int idx = blockIdx.x * blockDim.x + threadIdx.x;
    int total = M * (N_HEADS + N_KV) * (HD / 2);
    if (idx >= total) return;
    int j = idx & 31;
    int rest = idx >> 5;
    int h = rest % (N_HEADS + N_KV);
    int m = rest / (N_HEADS + N_KV);
    float t = tt[m];
    float inv = __powf(10000.f, -(2.f * (float)j) / (float)HD);
    float a = t * inv;
    float cs = __cosf(a), sn = __sinf(a);
    float* base = (h < N_HEADS) ? (q + ((size_t)m * N_HEADS + h) * HD)
                                : (k + ((size_t)m * N_KV + (h - N_HEADS)) * HD);
    float x1 = base[j], x2 = base[j + 32];
    base[j]      = x1 * cs - x2 * sn;
    base[j + 32] = x2 * cs + x1 * sn;
}

// ---------------------------------------------------------------------------
// Attention: grid (ceil(S/16), H, BD); 256 thr = 16 q-rows x 16 lanes.
// Scores kept in LDS; softmax; weighted V; writes bf16 for the wo GEMM.
// ---------------------------------------------------------------------------
__global__ __launch_bounds__(256)
void k_attn(const float* __restrict__ q, const float* __restrict__ k,
            const float* __restrict__ v, const float* __restrict__ madd,
            bf16* __restrict__ outb, int BD, int S) {
    __shared__ float qs[16][HD];
    __shared__ float sc[16][S_A];
    __shared__ float red[16][16];
    const int tx = threadIdx.x & 15, ty = threadIdx.x >> 4;
    const int qb = blockIdx.x, h = blockIdx.y, bd = blockIdx.z;
    const int kvh = h / (N_HEADS / N_KV);
    const int q_s = qb * 16 + ty;
    const bool qv = (q_s < S);

    {
        const float* src = q + ((size_t)(bd * S + (qv ? q_s : 0)) * (N_HEADS * HD)) + h * HD;
        for (int i = 0; i < 4; ++i)
            qs[ty][tx * 4 + i] = qv ? src[tx * 4 + i] : 0.f;
    }
    __syncthreads();

    for (int sb = 0; sb < S; sb += 16) {
        int s = sb + tx;
        if (s < S) {
            const float* kr = k + ((size_t)(bd * S + s) * (N_KV * HD)) + kvh * HD;
            float d = 0.f;
            for (int t = 0; t < HD; ++t) d += qs[ty][t] * kr[t];
            sc[ty][s] = d * 0.125f + (madd ? madd[(size_t)bd * S + s] : 0.f);
        }
    }
    __syncthreads();

    float pm = -3.4e38f;
    for (int s = tx; s < S; s += 16) pm = fmaxf(pm, sc[ty][s]);
    red[ty][tx] = pm; __syncthreads();
    float mx = red[ty][0];
    for (int i = 1; i < 16; ++i) mx = fmaxf(mx, red[ty][i]);
    __syncthreads();
    float ps = 0.f;
    for (int s = tx; s < S; s += 16) { float e = __expf(sc[ty][s] - mx); sc[ty][s] = e; ps += e; }
    red[ty][tx] = ps; __syncthreads();
    float sum = 0.f;
    for (int i = 0; i < 16; ++i) sum += red[ty][i];
    float inv = 1.f / sum;
    for (int s = tx; s < S; s += 16) sc[ty][s] *= inv;
    __syncthreads();

    if (qv) {
        float o0 = 0, o1 = 0, o2 = 0, o3 = 0;
        const float* vb = v + ((size_t)bd * S) * (N_KV * HD) + kvh * HD + tx * 4;
        for (int s = 0; s < S; ++s) {
            float w = sc[ty][s];
            const float* vr = vb + (size_t)s * (N_KV * HD);
            o0 += w * vr[0]; o1 += w * vr[1]; o2 += w * vr[2]; o3 += w * vr[3];
        }
        size_t ob = ((size_t)(bd * S + q_s)) * (N_HEADS * HD) + h * HD + tx * 4;
        outb[ob + 0] = (bf16)o0; outb[ob + 1] = (bf16)o1;
        outb[ob + 2] = (bf16)o2; outb[ob + 3] = (bf16)o3;
    }
}

// ---------------------------------------------------------------------------
// Masked mean pool over time + feature LayerNorm + var-embedding add.
// ---------------------------------------------------------------------------
__global__ __launch_bounds__(256)
void k_pool_ln(const float* __restrict__ xfin, const float* __restrict__ madd,
               const float* __restrict__ ln_g, const float* __restrict__ ln_b,
               const float* __restrict__ var_emb, float* __restrict__ xout) {
    int r = blockIdx.x;
    __shared__ float pooled[D_MODEL];
    __shared__ float red[256];
    float dn = 0.f;
    for (int s = threadIdx.x; s < L_OBS; s += 256)
        dn += (madd[(size_t)r * S_A + PROMPT + s] == 0.f) ? 1.f : 0.f;
    red[threadIdx.x] = dn; __syncthreads();
    for (int st = 128; st > 0; st >>= 1) {
        if (threadIdx.x < st) red[threadIdx.x] += red[threadIdx.x + st];
        __syncthreads();
    }
    float denom = red[0] + 1e-8f;
    __syncthreads();

    for (int c = threadIdx.x; c < D_MODEL; c += 256) {
        float acc = 0.f;
        for (int s = 0; s < L_OBS; ++s) {
            float m = (madd[(size_t)r * S_A + PROMPT + s] == 0.f) ? 1.f : 0.f;
            acc += xfin[((size_t)r * S_A + PROMPT + s) * D_MODEL + c] * m;
        }
        pooled[c] = acc / denom;
    }
    __syncthreads();

    float pmv = 0.f;
    for (int c = threadIdx.x; c < D_MODEL; c += 256) pmv += pooled[c];
    red[threadIdx.x] = pmv; __syncthreads();
    for (int st = 128; st > 0; st >>= 1) {
        if (threadIdx.x < st) red[threadIdx.x] += red[threadIdx.x + st];
        __syncthreads();
    }
    float mu = red[0] / (float)D_MODEL;
    __syncthreads();
    float pv = 0.f;
    for (int c = threadIdx.x; c < D_MODEL; c += 256) {
        float d = pooled[c] - mu; pv += d * d;
    }
    red[threadIdx.x] = pv; __syncthreads();
    for (int st = 128; st > 0; st >>= 1) {
        if (threadIdx.x < st) red[threadIdx.x] += red[threadIdx.x + st];
        __syncthreads();
    }
    float inv = rsqrtf(red[0] / (float)D_MODEL + 1e-5f);
    int dv = r % D_VARS;
    for (int c = threadIdx.x; c < D_MODEL; c += 256)
        xout[(size_t)r * D_MODEL + c] =
            (pooled[c] - mu) * inv * ln_g[c] + ln_b[c] + var_emb[(size_t)dv * D_MODEL + c];
}

__global__ void k_iota_mod(float* __restrict__ t, int n, int mod) {
    int i = blockIdx.x * blockDim.x + threadIdx.x;
    if (i < n) t[i] = (float)(i % mod);
}

// out[b, l, d] = fc[(b*12+d)*12 + 0], broadcast over l
__global__ void k_final(const float* __restrict__ fc, float* __restrict__ out) {
    int i = blockIdx.x * blockDim.x + threadIdx.x;
    if (i >= BATCH * L_PRED * D_VARS) return;
    int d = i % D_VARS;
    int b = i / (D_VARS * L_PRED);
    out[i] = fc[(size_t)(b * D_VARS + d) * D_VARS + 0];
}

// ---------------------------------------------------------------------------
// Host orchestration
// ---------------------------------------------------------------------------
static inline void launch_gemm(hipStream_t st, const bf16* A, const bf16* W,
                               const float* bias, const float* res, const float* aux,
                               float* Cf, bf16* Cb, int M, int N, int K, int mode) {
    dim3 g((N + GT_N - 1) / GT_N, (M + GT_M - 1) / GT_M);
    k_gemm<<<g, 256, 0, st>>>(A, W, bias, res, aux, Cf, Cb, M, N, K, mode);
}
static inline void launch_f2b(hipStream_t st, const float* in, bf16* out, size_t n) {
    k_f2b<<<(unsigned)((n + 255) / 256), 256, 0, st>>>(in, out, n);
}

extern "C" void kernel_launch(void* const* d_in, const int* in_sizes, int n_in,
                              void* d_out, int out_size, void* d_ws, size_t ws_size,
                              hipStream_t stream) {
    const float* observed_data = (const float*)d_in[0];
    const float* observed_mask = (const float*)d_in[1];
    const float* observed_tp   = (const float*)d_in[2];
    const float* val_w   = (const float*)d_in[4];
    const float* val_b   = (const float*)d_in[5];
    const float* prompt  = (const float*)d_in[6];
    const float* var_emb = (const float*)d_in[7];
    const float* ln_g    = (const float*)d_in[8];
    const float* ln_b    = (const float*)d_in[9];
    const float* head_w1 = (const float*)d_in[10];
    const float* head_b1 = (const float*)d_in[11];
    const float* head_w2 = (const float*)d_in[12];
    const float* head_b2 = (const float*)d_in[13];
    const float* wq = (const float*)d_in[14];
    const float* bq = (const float*)d_in[15];
    const float* wk = (const float*)d_in[16];
    const float* bk = (const float*)d_in[17];
    const float* wv = (const float*)d_in[18];
    const float* bv = (const float*)d_in[19];
    const float* wo = (const float*)d_in[20];
    const float* n1w = (const float*)d_in[21];
    const float* n2w = (const float*)d_in[22];
    const float* gate_w = (const float*)d_in[23];
    const float* up_w   = (const float*)d_in[24];
    const float* down_w = (const float*)d_in[25];
    const float* fnorm  = (const float*)d_in[26];
    float* out = (float*)d_out;
    (void)in_sizes; (void)n_in; (void)out_size;

    // ---- workspace carve-up (bump allocator) ----
    char* wsp = (char*)d_ws;
    size_t used = 0;
    auto alloc = [&](size_t bytes) -> void* {
        void* p = wsp + used;
        used += (bytes + 255) & ~(size_t)255;
        return p;
    };
    float* x     = (float*)alloc((size_t)M_A * D_MODEL * 4);   // hidden state (f32)
    bf16*  xn    = (bf16*)alloc((size_t)M_A * D_MODEL * 2);    // normed, bf16
    float* qbuf  = (float*)alloc((size_t)M_A * D_MODEL * 4);   // q / g-chunk / stageA final
    float* kbuf  = (float*)alloc((size_t)M_A * N_KV * HD * 4);
    float* vbuf  = (float*)alloc((size_t)M_A * N_KV * HD * 4);
    bf16*  attnb = (bf16*)alloc((size_t)M_A * D_MODEL * 2);    // attn out / e-chunk
    float* tt    = (float*)alloc((size_t)M_A * 4);
    float* madd  = (float*)alloc((size_t)M_A * 4);
    float* tt2   = (float*)alloc((size_t)M_B * 4);
    bf16*  wA    = (bf16*)alloc((size_t)D_MODEL * FF * 2);     // 8 MB weight slot
    bf16*  wB    = (bf16*)alloc((size_t)D_MODEL * FF * 2);
    bf16*  wC    = (bf16*)alloc((size_t)D_MODEL * FF * 2);
    bf16*  hhead = (bf16*)alloc((size_t)M_B * (D_MODEL / 2) * 2);
    float* fc    = (float*)alloc((size_t)M_B * D_VARS * 4);
    if (used > ws_size) return;  // scratch too small; nothing sane to do

    const size_t QKW = (size_t)D_MODEL * D_MODEL;   // wq / wo
    const size_t KVW = (size_t)D_MODEL * (N_KV * HD);
    const size_t FFW = (size_t)D_MODEL * FF;

    // ================= Stage A: series transformer =================
    k_build_inputs<<<M_A, 256, 0, stream>>>(observed_data, observed_mask, observed_tp,
                                            val_w, val_b, prompt, x, tt, madd);
    for (int stg = 0; stg < 2; ++stg) {
        const int Ms = (stg == 0) ? M_A : M_B;
        const int Ss = (stg == 0) ? S_A : S_B;
        const int BDs = (stg == 0) ? (BATCH * D_VARS) : BATCH;
        const float* tvec = (stg == 0) ? tt : tt2;
        const float* mvec = (stg == 0) ? madd : nullptr;
        if (stg == 1)
            k_iota_mod<<<1, 64, 0, stream>>>(tt2, M_B, D_VARS);

        for (int l = 0; l < 2; ++l) {
            const int sl = stg * 2 + l;
            // ---- attention ----
            launch_f2b(stream, wq + (size_t)sl * QKW, wA, QKW);
            launch_f2b(stream, wk + (size_t)sl * KVW, wB, KVW);
            launch_f2b(stream, wv + (size_t)sl * KVW, wC, KVW);
            k_rmsnorm<<<Ms, 256, 0, stream>>>(x, n1w + (size_t)sl * D_MODEL, xn, nullptr, D_MODEL);
            launch_gemm(stream, xn, wA, bq + (size_t)sl * D_MODEL, nullptr, nullptr,
                        qbuf, nullptr, Ms, D_MODEL, D_MODEL, 0);
            launch_gemm(stream, xn, wB, bk + (size_t)sl * (N_KV * HD), nullptr, nullptr,
                        kbuf, nullptr, Ms, N_KV * HD, D_MODEL, 0);
            launch_gemm(stream, xn, wC, bv + (size_t)sl * (N_KV * HD), nullptr, nullptr,
                        vbuf, nullptr, Ms, N_KV * HD, D_MODEL, 0);
            {
                int total = Ms * (N_HEADS + N_KV) * (HD / 2);
                k_rope<<<(total + 255) / 256, 256, 0, stream>>>(qbuf, kbuf, tvec, Ms);
            }
            {
                dim3 g((Ss + 15) / 16, N_HEADS, BDs);
                k_attn<<<g, 256, 0, stream>>>(qbuf, kbuf, vbuf, mvec, attnb, BDs, Ss);
            }
            launch_f2b(stream, wo + (size_t)sl * QKW, wA, QKW);
            launch_gemm(stream, attnb, wA, nullptr, x, nullptr, x, nullptr,
                        Ms, D_MODEL, D_MODEL, 0);          // residual in place
            // ---- MLP (chunked; g aliases qbuf, e aliases attnb) ----
            k_rmsnorm<<<Ms, 256, 0, stream>>>(x, n2w + (size_t)sl * D_MODEL, xn, nullptr, D_MODEL);
            launch_f2b(stream, gate_w + (size_t)sl * FFW, wA, FFW);
            launch_f2b(stream, up_w   + (size_t)sl * FFW, wB, FFW);
            launch_f2b(stream, down_w + (size_t)sl * FFW, wC, FFW);
            float* gch = qbuf;
            bf16*  ech = attnb;
            for (int c0 = 0; c0 < Ms; c0 += MLP_CHUNK) {
                int rows = (Ms - c0 < MLP_CHUNK) ? (Ms - c0) : MLP_CHUNK;
                launch_gemm(stream, xn + (size_t)c0 * D_MODEL, wA, nullptr, nullptr, nullptr,
                            gch, nullptr, rows, FF, D_MODEL, 0);
                launch_gemm(stream, xn + (size_t)c0 * D_MODEL, wB, nullptr, nullptr, gch,
                            nullptr, ech, rows, FF, D_MODEL, 1);        // silu(g)*u -> bf16
                launch_gemm(stream, ech, wC, nullptr, x + (size_t)c0 * D_MODEL, nullptr,
                            x + (size_t)c0 * D_MODEL, nullptr, rows, D_MODEL, FF, 0);
            }
        }

        if (stg == 0) {
            // final rmsnorm of stage A (f32 into qbuf), pool+LN -> var_inputs in x[0..23]
            k_rmsnorm<<<M_A, 256, 0, stream>>>(x, fnorm + 0, nullptr, qbuf, D_MODEL);
            k_pool_ln<<<M_B, 256, 0, stream>>>(qbuf, madd, ln_g, ln_b, var_emb, x);
        }
    }

    // ================= Head =================
    k_rmsnorm<<<M_B, 256, 0, stream>>>(x, fnorm + D_MODEL, xn, nullptr, D_MODEL);
    launch_f2b(stream, head_w1, wA, (size_t)D_MODEL * (D_MODEL / 2));
    launch_gemm(stream, xn, wA, head_b1, nullptr, nullptr, nullptr, hhead,
                M_B, D_MODEL / 2, D_MODEL, 2);             // leaky relu
    launch_f2b(stream, head_w2, wB, (size_t)(D_MODEL / 2) * D_VARS);
    launch_gemm(stream, hhead, wB, head_b2, nullptr, nullptr, fc, nullptr,
                M_B, D_VARS, D_MODEL / 2, 0);
    k_final<<<(BATCH * L_PRED * D_VARS + 255) / 256, 256, 0, stream>>>(fc, out);
}
